// SMB_27032524161691
// MI455X (gfx1250) — matrix operand order repaired and checked by reading, not compiled, not run on hardware
//
#include <hip/hip_runtime.h>
#include <math.h>

// MI455X (gfx1250) wave32 / WMMA implementation.
// Roofline: ~290 GFLOP, ~1 GB HBM traffic -> f16 WMMA (16x16x32, f32 acc)
// balances compute (~58us) against 23.3 TB/s memory (~50us). fp32 WMMA
// (16x16x4) would be ~8x compute bound, so activations/weights are f16.
//
// GEMM roles: A = weights (M = out-channels), B = activations (N = pixels).
// D-tile layout (lane = N index, VGPR r = M row) => each lane owns 8
// consecutive out-channels of ONE pixel => packed b128 epilogue stores.
// Weight fragments are loaded transiently (one at a time) to keep live
// VGPRs ~160 (128 acc + 16 bf + 8 wf) and avoid scratch spills.

typedef _Float16 half_t;
typedef __attribute__((ext_vector_type(8)))  _Float16 v8h;
typedef __attribute__((ext_vector_type(16))) _Float16 v16h;
typedef __attribute__((ext_vector_type(8)))  float    v8f;

#define HDIM 512
#define WDIM 512
#define CCH  64
#define NLAY 4

static __device__ __forceinline__ v16h cat8(v8h lo, v8h hi) {
  return __builtin_shufflevector(lo, hi, 0,1,2,3,4,5,6,7,8,9,10,11,12,13,14,15);
}
static __device__ __forceinline__ v8f zero8f() {
  v8f z = {0.f,0.f,0.f,0.f,0.f,0.f,0.f,0.f}; return z;
}
static __device__ __forceinline__ v8h zero8h() {
  v8h z;
#pragma unroll
  for (int i = 0; i < 8; ++i) z[i] = (_Float16)0.0f;
  return z;
}

// ---------------------------------------------------------------------------
// Gumbel-softmax channel routing mask: ch (1,L,2,C) -> written to d_out tail.
// ---------------------------------------------------------------------------
__global__ void k_prep_ch(const float* __restrict__ gum,
                          const float* __restrict__ msk,
                          float* __restrict__ chOut) {
  int c = threadIdx.x;
  if (c >= CCH) return;
  for (int l = 0; l < NLAY; ++l) {
    float u0 = gum[l*2*CCH + c];
    float u1 = gum[l*2*CCH + CCH + c];
    u0 = fminf(fmaxf(u0, 1e-10f), 1.0f);
    u1 = fminf(fmaxf(u1, 1e-10f), 1.0f);
    float a0 = msk[l*2*CCH + c]       + (-logf(-logf(u0)));   // TAU = 1
    float a1 = msk[l*2*CCH + CCH + c] + (-logf(-logf(u1)));
    float m  = fmaxf(a0, a1);
    float e0 = expf(a0 - m), e1 = expf(a1 - m);
    float inv = 1.0f / (e0 + e1);
    chOut[l*128 + c]       = e0 * inv;   // cd (dense)
    chOut[l*128 + 64 + c]  = e1 * inv;   // cs (sparse)
  }
}

// ---------------------------------------------------------------------------
// Fold channel masks into f16 weights: wT[slot][tap][co][ci].
// slot0: w0 unscaled. slot 2l-1 / 2l: w_l scaled by cd_{l-1}/cs_{l-1} on ci.
// Also wcT[co][l*64+ci] (f16 transposed w_collect).
// ---------------------------------------------------------------------------
__global__ void k_prep_w(const float* __restrict__ w0, const float* __restrict__ w1,
                         const float* __restrict__ w2, const float* __restrict__ w3,
                         const float* __restrict__ wc, const float* __restrict__ chv,
                         half_t* __restrict__ wT, half_t* __restrict__ wcT) {
  const int NW = 7 * 9 * 64 * 64;  // 258048
  const int NT = NW + 64 * 256;
  for (int idx = blockIdx.x * blockDim.x + threadIdx.x; idx < NT;
       idx += gridDim.x * blockDim.x) {
    if (idx < NW) {
      int slot = idx / 36864;
      int r    = idx - slot * 36864;
      int tap  = r >> 12;
      int co   = (r >> 6) & 63;
      int ci   = r & 63;
      const float* w; float scale = 1.0f;
      if (slot == 0) { w = w0; }
      else {
        int l = (slot + 1) >> 1;                // 1,2->1  3,4->2  5,6->3
        w = (l == 1) ? w1 : (l == 2) ? w2 : w3;
        int s = (slot & 1) ? 0 : 1;             // odd slot = dense set
        scale = chv[(l - 1) * 128 + s * 64 + ci];
      }
      wT[idx] = (half_t)(w[(tap << 12) + (ci << 6) + co] * scale);
    } else {
      int j   = idx - NW;
      int co  = j >> 8;
      int cik = j & 255;
      wcT[j] = (half_t)(wc[cik * 64 + co]);
    }
  }
}

// ---------------------------------------------------------------------------
// x0 f32 -> f16
// ---------------------------------------------------------------------------
__global__ void k_cvt(const float* __restrict__ x, half_t* __restrict__ xh, int n) {
  int i = (blockIdx.x * blockDim.x + threadIdx.x) * 4;
  if (i >= n) return;
  float4 v = *(const float4*)(x + i);
  xh[i]   = (half_t)v.x; xh[i+1] = (half_t)v.y;
  xh[i+2] = (half_t)v.z; xh[i+3] = (half_t)v.w;
}

// ---------------------------------------------------------------------------
// Fused conv3x3 (implicit GEMM, f16 WMMA, f32 acc) + mask epilogue + relu.
// WG = 256 thr (8 waves): 16x16 pixel tile x 64 out-ch.
// A = weights (M=co, 4 tiles), B = activations (N=pixels, wave owns rows
// {wave, wave+8}). SETS=1 (layer 0) or SETS=2 (dense+sparse weight sets).
// LDS: 18x18x64 f16 halo + SETS*9*64*64 f16 weights + spa tile + ch vector.
// ---------------------------------------------------------------------------
template<int SETS>
__global__ void k_conv(const half_t* __restrict__ src,   // (B,H,W,64) f16
                       const half_t* __restrict__ wsrc,  // SETS*[tap][co][ci] f16
                       const float*  __restrict__ spa,   // (B,H,W,1)
                       const float*  __restrict__ chL,   // cd[64], cs[64]
                       half_t* __restrict__ dst) {
  extern __shared__ char smem[];
  half_t* aT  = (half_t*)smem;                      // 18*18*64 halves
  half_t* wS  = aT + 18 * 18 * 64;                  // up to 2*36864 halves
  float*  sT  = (float*)(smem + (18 * 18 * 64 + 2 * 36864) * 2); // 256 f32
  float*  chS = sT + 256;                           // 128 f32

  const int tid  = threadIdx.x;
  const int lane = tid & 31;
  const int wave = tid >> 5;
  const int bx = blockIdx.x, by = blockIdx.y, bb = blockIdx.z;

  // stage activations (zero-padded halo)
  for (int idx = tid; idx < 18 * 18 * 8; idx += 256) {
    int pix = idx >> 3, part = idx & 7;
    int iy = pix / 18, ix = pix - iy * 18;
    int gy = by * 16 + iy - 1, gx = bx * 16 + ix - 1;
    v8h v = zero8h();
    if ((unsigned)gy < (unsigned)HDIM && (unsigned)gx < (unsigned)WDIM)
      v = *(const v8h*)(src + ((((size_t)bb * HDIM + gy) * WDIM + gx) << 6) + part * 8);
    *(v8h*)(aT + pix * 64 + part * 8) = v;
  }
  // stage pre-scaled weights
  for (int idx = tid; idx < SETS * 36864 / 8; idx += 256)
    *(v8h*)(wS + idx * 8) = *(const v8h*)(wsrc + idx * 8);
  // stage spatial mask tile + channel mask vector
  {
    int pr = tid >> 4, pcx = tid & 15;
    sT[tid] = spa[((size_t)bb * HDIM + by * 16 + pr) * WDIM + bx * 16 + pcx];
    if (tid < 128) chS[tid] = chL[tid];
  }
  __syncthreads();

  v8f acc[2][4][SETS];                // [pix-row group][co tile][set]
#pragma unroll
  for (int g = 0; g < 2; ++g)
#pragma unroll
    for (int m = 0; m < 4; ++m)
#pragma unroll
      for (int s = 0; s < SETS; ++s) acc[g][m][s] = zero8f();

  const int pc = lane & 15;     // A: co row in M-tile / B: pixel col in N-tile
  const int hi = lane >> 4;
  const int aoff = hi << 4;     // A K-halves: 0..15 / 16..31
  const int boff = hi << 4;     // B K-halves: 0..15 / 16..31

#pragma unroll
  for (int tap = 0; tap < 9; ++tap) {
    const int ky = tap / 3, kx = tap - ky * 3;
#pragma unroll
    for (int c = 0; c < 2; ++c) {               // 32-channel K chunks
      // activation fragments (B) for this K step: both pixel-row groups
      v16h bf[2];
#pragma unroll
      for (int g = 0; g < 2; ++g) {
        int pr = wave + g * 8;
        const half_t* bp = aT + ((pr + ky) * 18 + (pc + kx)) * 64 + c * 32 + boff;
        bf[g] = cat8(*(const v8h*)bp, *(const v8h*)(bp + 8));
      }
      // weight fragments (A) loaded transiently, each feeding 2 WMMAs
#pragma unroll
      for (int m = 0; m < 4; ++m) {
#pragma unroll
        for (int s = 0; s < SETS; ++s) {
          const half_t* ap = wS + s * 36864 + (tap * 64 + (m * 16 + pc)) * 64
                               + c * 32 + aoff;
          v16h wf = cat8(*(const v8h*)ap, *(const v8h*)(ap + 8));
#pragma unroll
          for (int g = 0; g < 2; ++g)
            acc[g][m][s] = __builtin_amdgcn_wmma_f32_16x16x32_f16(
                false, wf, false, bf[g], (short)0, acc[g][m][s], false, false);
        }
      }
    }
  }

  // epilogue: out = relu(Fd*(cd + cs*spa) [+ Fs*spa])   (cd+cs == 1)
  // D layout: lane pc = pixel, VGPR r -> co = m*16 + hi*8 + r  => packed b128.
#pragma unroll
  for (int m = 0; m < 4; ++m) {
    const int cob = m * 16 + hi * 8;
    float cdv[8], csv[8];
    *(float4*)(cdv)     = *(const float4*)(chS + cob);
    *(float4*)(cdv + 4) = *(const float4*)(chS + cob + 4);
    *(float4*)(csv)     = *(const float4*)(chS + 64 + cob);
    *(float4*)(csv + 4) = *(const float4*)(chS + 64 + cob + 4);
#pragma unroll
    for (int g = 0; g < 2; ++g) {
      const int pr = wave + g * 8;
      const float sp = sT[pr * 16 + pc];
      v8h hv;
#pragma unroll
      for (int r = 0; r < 8; ++r) {
        float v = acc[g][m][0][r] * (cdv[r] + csv[r] * sp);
        if constexpr (SETS == 2) v += acc[g][m][1][r] * sp;
        v = v > 0.f ? v : 0.f;
        hv[r] = (half_t)v;
      }
      size_t gp = ((size_t)bb * HDIM + by * 16 + pr) * WDIM + bx * 16 + pc;
      *(v8h*)(dst + (gp << 6) + cob) = hv;
    }
  }
}

// ---------------------------------------------------------------------------
// Collect: out[p,co] = sum_{l,ci} feat_l[p,ci] * wc[l*64+ci,co] + b[co]
// WMMA GEMM: A = wcT (M=co), B = feats (N=pixels). WG: 128 pixels.
// ---------------------------------------------------------------------------
__global__ void k_collect(const half_t* __restrict__ f0, const half_t* __restrict__ f1,
                          const half_t* __restrict__ f2, const half_t* __restrict__ f3,
                          const half_t* __restrict__ wcT,   // [co][256] f16
                          const float* __restrict__ bc,
                          float* __restrict__ out) {
  __shared__ half_t wS[64 * 256];
  __shared__ float  sB[64];
  const int tid = threadIdx.x, lane = tid & 31, wave = tid >> 5;
  for (int idx = tid; idx < 64 * 256 / 8; idx += 256)
    *(v8h*)(wS + idx * 8) = *(const v8h*)(wcT + idx * 8);
  if (tid < 64) sB[tid] = bc[tid];
  __syncthreads();

  const half_t* feats[4] = {f0, f1, f2, f3};
  const int pc = lane & 15, hi = lane >> 4;
  const int aoff = hi << 4, boff = hi << 4;
  const size_t p = (size_t)blockIdx.x * 128 + wave * 16 + pc;  // this lane's pixel

  v8f acc[4];
#pragma unroll
  for (int m = 0; m < 4; ++m) acc[m] = zero8f();

#pragma unroll
  for (int l = 0; l < 4; ++l) {
    const half_t* fp = feats[l];
#pragma unroll
    for (int c = 0; c < 2; ++c) {
      const half_t* bp = fp + (p << 6) + c * 32 + boff;       // activations (B)
      v16h b = cat8(*(const v8h*)bp, *(const v8h*)(bp + 8));
#pragma unroll
      for (int m = 0; m < 4; ++m) {
        const half_t* ap = wS + (m * 16 + pc) * 256 + l * 64 + c * 32 + aoff;
        v16h a = cat8(*(const v8h*)ap, *(const v8h*)(ap + 8));
        acc[m] = __builtin_amdgcn_wmma_f32_16x16x32_f16(
            false, a, false, b, (short)0, acc[m], false, false);
      }
    }
  }
  // lane holds co = m*16 + hi*8 + r for its pixel -> two float4 stores / tile
#pragma unroll
  for (int m = 0; m < 4; ++m) {
    const int cob = m * 16 + hi * 8;
    float4 o0, o1;
    o0.x = acc[m][0] + sB[cob + 0]; o0.y = acc[m][1] + sB[cob + 1];
    o0.z = acc[m][2] + sB[cob + 2]; o0.w = acc[m][3] + sB[cob + 3];
    o1.x = acc[m][4] + sB[cob + 4]; o1.y = acc[m][5] + sB[cob + 5];
    o1.z = acc[m][6] + sB[cob + 6]; o1.w = acc[m][7] + sB[cob + 7];
    *(float4*)(out + (p << 6) + cob)     = o0;
    *(float4*)(out + (p << 6) + cob + 4) = o1;
  }
}

// ---------------------------------------------------------------------------
extern "C" void kernel_launch(void* const* d_in, const int* in_sizes, int n_in,
                              void* d_out, int out_size, void* d_ws, size_t ws_size,
                              hipStream_t stream) {
  const float* x0  = (const float*)d_in[0];
  const float* spa = (const float*)d_in[1];
  const float* gum = (const float*)d_in[2];
  const float* msk = (const float*)d_in[3];
  const float* w0  = (const float*)d_in[4];
  const float* w1  = (const float*)d_in[5];
  const float* w2  = (const float*)d_in[6];
  const float* w3  = (const float*)d_in[7];
  const float* wc  = (const float*)d_in[8];
  const float* bc  = (const float*)d_in[9];
  float* out = (float*)d_out;

  const size_t NPIX = (size_t)2 * HDIM * WDIM;   // 524288
  const size_t FEAT = NPIX * CCH;                // 33554432 elements

  // workspace layout (~336 MB): x0h | feat0..3 (f16) | wT blobs | wcT
  char* ws = (char*)d_ws;
  half_t* x0h = (half_t*)ws;
  half_t* feat[4];
  for (int l = 0; l < 4; ++l) feat[l] = (half_t*)(ws + FEAT * 2 * (size_t)(1 + l));
  half_t* wT  = (half_t*)(ws + FEAT * 2 * 5);
  half_t* wcT = wT + 7 * 36864;
  float*  chOut = out + FEAT;                    // ch goes to d_out tail

  k_prep_ch<<<1, 64, 0, stream>>>(gum, msk, chOut);
  k_prep_w<<<256, 256, 0, stream>>>(w0, w1, w2, w3, wc, chOut, wT, wcT);
  k_cvt<<<(int)(FEAT / 4 / 256), 256, 0, stream>>>(x0, x0h, (int)FEAT);

  dim3 grid(WDIM / 16, HDIM / 16, 2);
  size_t lds = (size_t)(18 * 18 * 64 + 2 * 36864) * 2 + 256 * 4 + 128 * 4; // 190,464 B
  k_conv<1><<<grid, 256, lds, stream>>>(x0h,     wT + 0 * 36864, spa, chOut + 0 * 128, feat[0]);
  k_conv<2><<<grid, 256, lds, stream>>>(feat[0], wT + 1 * 36864, spa, chOut + 1 * 128, feat[1]);
  k_conv<2><<<grid, 256, lds, stream>>>(feat[1], wT + 3 * 36864, spa, chOut + 2 * 128, feat[2]);
  k_conv<2><<<grid, 256, lds, stream>>>(feat[2], wT + 5 * 36864, spa, chOut + 3 * 128, feat[3]);

  k_collect<<<(int)(NPIX / 128), 256, 0, stream>>>(feat[0], feat[1], feat[2], feat[3],
                                                   wcT, bc, out);
  (void)in_sizes; (void)n_in; (void)out_size; (void)ws_size;
}
